// CrossPartCoordinator_52295521796651
// MI455X (gfx1250) — compile-verified, matrix operand order checked by reading, not run on hardware
//
#include <hip/hip_runtime.h>
#include <hip/hip_bf16.h>

#define DEVI __device__ __forceinline__

typedef __attribute__((ext_vector_type(16))) __bf16 bf16x16;
typedef __attribute__((ext_vector_type(8)))  float  f32x8;

static constexpr int NP = 6, D = 256, H = 4, DH = 64;
static constexpr int Tt = 1024, Mtok = 32 * 1024;   // B*T tokens
static constexpr int DPAD = 96;                     // per-part input dim padded to 3 k-steps
static constexpr int TOK = 8;                       // tokens per K1 block (48 attn rows = 3 m-tiles)

constexpr int PDIM[NP] = {8, 64, 64, 96, 64, 64};
// kinematic mask rows (bit kp set => attend allowed)
constexpr int KIN[NP] = {0x3F, 0x0B, 0x0D, 0x3F, 0x19, 0x29};

struct Parts {
  const float* z[NP];
  const float* wp[NP];
  const float* bp[NP];
};

DEVI __bf16 tobf(float x) { return (__bf16)x; }
DEVI float  fromb(__bf16 x) { return (float)x; }

DEVI f32x8 wmma_bf16(bf16x16 a, bf16x16 b, f32x8 c) {
  // v_wmma_f32_16x16x32_bf16: D = A(16x32) * B(32x16) + C(16x16 f32)
  return __builtin_amdgcn_wmma_f32_16x16x32_bf16(false, a, false, b, (short)0, c, false, false);
}
DEVI bf16x16 fragp(const __bf16* p) { return *(const bf16x16*)p; }

// Offset of element (row mloc, k klocal) inside a 512-elem 16x32 bf16 fragment stored
// as [lane][16 halves] per the ISA A-matrix layout:
//   k = (e&7) + ((e>>3)<<4) + ((lane>>4)<<3), m = lane&15
DEVI int frag_off(int mloc, int klocal) {
  int laneH = (klocal >> 3) & 1;
  int e = (klocal & 7) | (((klocal >> 4) & 1) << 3);
  return ((mloc & 15) + (laneH << 4)) * 16 + e;
}

// ---------------------------------------------------------------------------
// K0: weight prep. Builds (all bf16, pre-swizzled to WMMA fragment order):
//   WtB  [3][6][nt=16][kt=3][512] : combined QKV weights  B[k=i][n=o] = (in_proj @ w_proj[p])
//   biasC[3][6][256]  f32         : combined QKV biases
//   owtB [nt=16][kt=8][512]       : out-proj  B[k][n] = out_w[n][k]
//   cwtB [6][tap=3][nt=16][kt=8][512] : conv  B[k][n] = conv_w[p*256+n][k][tap]
// ---------------------------------------------------------------------------
__global__ void k0_prep(Parts prt,
                        const float* __restrict__ ipw, const float* __restrict__ ipb,
                        const float* __restrict__ ow,  const float* __restrict__ cw,
                        __bf16* __restrict__ WtB, float* __restrict__ biasC,
                        __bf16* __restrict__ owtB, __bf16* __restrict__ cwtB)
{
  constexpr long NA = 3L * NP * 16 * 3 * 512;   // 442368
  constexpr long NB = 3L * NP * D;              // 4608
  constexpr long NC = 16L * 8 * 512;            // 65536
  constexpr long ND = (long)NP * 3 * 16 * 8 * 512; // 1179648
  long i = (long)blockIdx.x * blockDim.x + threadIdx.x;

  if (i < NA) {
    int e = i & 15, lane = (int)((i >> 4) & 31);
    long u = i >> 9;
    int kt = (int)(u % 3); u /= 3;
    int nt = (int)(u % 16); u /= 16;
    int p = (int)(u % NP);
    int t = (int)(u / NP);                       // 0=q 1=k 2=v
    int n = nt * 16 + (lane & 15);
    int k = kt * 32 + (e & 7) + ((e >> 3) << 4) + ((lane >> 4) << 3);
    float v = 0.f;
    if (k < PDIM[p]) {
      const float* wr = ipw + (long)(t * D + n) * D; // row of in_proj (length 256)
      const float* wc = prt.wp[p] + k;               // column of w_proj[p], stride d_p
      int dp = PDIM[p];
      float s = 0.f;
      for (int o = 0; o < D; ++o) s += wr[o] * wc[(long)o * dp];
      v = s;
    }
    WtB[i] = tobf(v);
    return;
  }
  i -= NA;
  if (i < NB) {
    int n = (int)(i & 255);
    int tp = (int)(i >> 8);
    int p = tp % NP, t = tp / NP;
    const float* wr = ipw + (long)(t * D + n) * D;
    float s = ipb[t * D + n];
    for (int o = 0; o < D; ++o) s += wr[o] * prt.bp[p][o];
    biasC[i] = s;
    return;
  }
  i -= NB;
  if (i < NC) {
    int e = (int)(i & 15), lane = (int)((i >> 4) & 31);
    int kt = (int)((i >> 9) & 7), nt = (int)(i >> 12);
    int n = nt * 16 + (lane & 15);
    int k = kt * 32 + (e & 7) + ((e >> 3) << 4) + ((lane >> 4) << 3);
    owtB[i] = tobf(ow[(long)n * D + k]);
    return;
  }
  i -= NC;
  if (i < ND) {
    int e = (int)(i & 15), lane = (int)((i >> 4) & 31);
    int kt = (int)((i >> 9) & 7), nt = (int)((i >> 12) & 15);
    int pt = (int)(i >> 16);
    int tau = pt % 3, p = pt / 3;
    int n = nt * 16 + (lane & 15);
    int k = kt * 32 + (e & 7) + ((e >> 3) << 4) + ((lane >> 4) << 3);
    cwtB[i] = tobf(cw[((long)(p * D + n) * D + k) * 3 + tau]);
  }
}

// ---------------------------------------------------------------------------
// K1: fused (folded) QKV projection -> masked 6x6 attention -> out-projection.
// One block = 8 tokens (48 part-rows). Writes attn_out fp32 to d_out and a
// bf16 copy to attnbuf for the conv kernel.
// ---------------------------------------------------------------------------
__global__ __launch_bounds__(256) void k1_qkv_attn(
    Parts prt,
    const __bf16* __restrict__ WtB, const float* __restrict__ biasC,
    const __bf16* __restrict__ owtB, const float* __restrict__ out_b,
    float* __restrict__ out, __bf16* __restrict__ attnbuf)
{
  __shared__ alignas(32) __bf16 zfrag[NP][3][512];        // z tile as A-fragments (rows 8..15 zero)
  __shared__ alignas(32) __bf16 qkvh[3][NP][TOK][DH];     // q/k/v head slice
  __shared__ alignas(32) __bf16 ofrag[3][8][512];         // attn output rows as A-fragments

  const int tid = threadIdx.x, lane = tid & 31, wv = tid >> 5;
  const int m0 = blockIdx.x * TOK;

  // ---- stage Z: z -> bf16 A-fragments in LDS (zero padded in k and rows) ----
  for (int idx = tid; idx < NP * 16 * DPAD; idx += 256) {
    int p = idx / (16 * DPAD);
    int r = idx % (16 * DPAD);
    int m = r / DPAD, k = r % DPAD;
    float v = 0.f;
    if (m < TOK && k < PDIM[p]) v = prt.z[p][(long)(m0 + m) * PDIM[p] + k];
    zfrag[p][k >> 5][frag_off(m, k & 31)] = tobf(v);
  }
  __syncthreads();

  for (int h = 0; h < H; ++h) {
    // ---- stage A: q/k/v head-slice GEMMs (K padded to 96, N = 64) ----
    for (int u = wv; u < 3 * NP * 4; u += 8) {
      int t = u / (NP * 4);
      int p = (u / 4) % NP;
      int ntl = u & 3;
      int nt = h * 4 + ntl;
      const __bf16* bbase = WtB + ((long)(((t * NP + p) * 16 + nt) * 3) << 9) + lane * 16;
      __builtin_prefetch(bbase, 0, 1);
      f32x8 acc = {};
      #pragma unroll
      for (int kt = 0; kt < 3; ++kt) {
        bf16x16 a = fragp(&zfrag[p][kt][lane * 16]);
        bf16x16 b = fragp(bbase + ((long)kt << 9));
        acc = wmma_bf16(a, b, acc);
      }
      int col = nt * 16 + (lane & 15);
      float bs = biasC[(t * NP + p) * D + col];
      int mh = (lane >> 4) << 3;
      #pragma unroll
      for (int r2 = 0; r2 < 8; ++r2) {
        int m = r2 + mh;
        if (m < TOK) qkvh[t][p][m][ntl * 16 + (lane & 15)] = tobf(acc[r2] + bs);
      }
    }
    __syncthreads();

    // ---- stage B: masked softmax attention for this head (48 rows, VALU) ----
    if (tid < TOK * NP) {
      int tok = tid / NP, qp = tid % NP;
      float qv[DH];
      #pragma unroll
      for (int i2 = 0; i2 < DH; ++i2) qv[i2] = fromb(qkvh[0][qp][tok][i2]);
      float sc[NP];
      int mrow = KIN[qp];
      float mx = -3.4e38f;
      for (int kp = 0; kp < NP; ++kp) {
        float s = 0.f;
        for (int i2 = 0; i2 < DH; ++i2) s += qv[i2] * fromb(qkvh[1][kp][tok][i2]);
        s *= 0.125f;                               // dh^-0.5, dh = 64
        if (!((mrow >> kp) & 1)) s = -3.0e38f;     // ~ finfo(f32).min
        sc[kp] = s;
        mx = fmaxf(mx, s);
      }
      float den = 0.f;
      for (int kp = 0; kp < NP; ++kp) { sc[kp] = __expf(sc[kp] - mx); den += sc[kp]; }
      float inv = 1.f / den;
      int orow = tok * NP + qp;                    // 0..47, matches output row m0*6+orow
      int mt = orow >> 4, mloc = orow & 15;
      for (int ch = 0; ch < DH; ++ch) {
        float ov = 0.f;
        for (int kp = 0; kp < NP; ++kp) ov += sc[kp] * fromb(qkvh[2][kp][tok][ch]);
        ov *= inv;
        int kg = h * DH + ch;
        ofrag[mt][kg >> 5][frag_off(mloc, kg & 31)] = tobf(ov);
      }
    }
    __syncthreads();
  }

  // ---- stage C: out-projection, 48 x 256, K = 256 (8 k-steps) ----
  const int nt0 = wv, nt1 = wv + 8;
  for (int mt = 0; mt < 3; ++mt) {
    f32x8 a0 = {}, a1 = {};
    #pragma unroll
    for (int kt = 0; kt < 8; ++kt) {
      bf16x16 a = fragp(&ofrag[mt][kt][lane * 16]);
      bf16x16 b0 = fragp(owtB + ((long)(nt0 * 8 + kt) << 9) + lane * 16);
      bf16x16 b1 = fragp(owtB + ((long)(nt1 * 8 + kt) << 9) + lane * 16);
      a0 = wmma_bf16(a, b0, a0);
      a1 = wmma_bf16(a, b1, a1);
    }
    int mh = (lane >> 4) << 3;
    int c0 = nt0 * 16 + (lane & 15);
    int c1 = nt1 * 16 + (lane & 15);
    float ob0 = out_b[c0], ob1 = out_b[c1];
    #pragma unroll
    for (int r2 = 0; r2 < 8; ++r2) {
      int m = mt * 16 + r2 + mh;                   // 0..47
      long gidx = ((long)m0 * NP + m) * D;
      float v0 = a0[r2] + ob0;
      float v1 = a1[r2] + ob1;
      out[gidx + c0] = v0;  attnbuf[gidx + c0] = tobf(v0);
      out[gidx + c1] = v1;  attnbuf[gidx + c1] = tobf(v1);
    }
  }
}

// ---------------------------------------------------------------------------
// K2: grouped temporal conv (k=3, pad 1) as 3-tap GEMM + residual add.
// One block = (batch b, 16 consecutive t, part p). Halo rows come from the
// bf16 attn buffer (never written here) -> race-free d_out += temporal.
// ---------------------------------------------------------------------------
__global__ __launch_bounds__(256) void k2_conv_add(
    const __bf16* __restrict__ attnbuf, const __bf16* __restrict__ cwtB,
    const float* __restrict__ conv_b, float* __restrict__ out)
{
  __shared__ alignas(32) __bf16 swz[3][8][512];  // per-tap A-fragments (16 rows x 256 k)
  const int tid = threadIdx.x, lane = tid & 31, wv = tid >> 5;
  int bid = blockIdx.x;
  int p = bid % NP;
  int m0 = (bid / NP) * 16;                      // first output token (tiles never cross batch)
  int b = m0 / Tt, t0 = m0 % Tt;

  for (int idx = tid; idx < 3 * 16 * D; idx += 256) {
    int tau = idx / (16 * D);
    int r = idx % (16 * D);
    int m = r / D, k = r % D;
    int t = t0 - 1 + tau + m;                    // conv input row for output t0+m, tap tau
    __bf16 v = tobf(0.f);
    if (t >= 0 && t < Tt) v = attnbuf[((long)(b * Tt + t) * NP + p) * D + k];
    swz[tau][k >> 5][frag_off(m, k & 31)] = v;
  }
  __syncthreads();

  const int nts[2] = {wv, wv + 8};
  f32x8 acc[2] = {{}, {}};
  for (int tau = 0; tau < 3; ++tau) {
    long base = ((long)(p * 3 + tau)) << 16;     // (p,tau) block: 16*8*512 elems
    __builtin_prefetch(cwtB + base + ((long)(nts[0] * 8) << 9), 0, 1);
    #pragma unroll
    for (int kt = 0; kt < 8; ++kt) {
      bf16x16 a = fragp(&swz[tau][kt][lane * 16]);
      #pragma unroll
      for (int s = 0; s < 2; ++s) {
        bf16x16 bf = fragp(cwtB + base + ((long)(nts[s] * 8 + kt) << 9) + lane * 16);
        acc[s] = wmma_bf16(a, bf, acc[s]);
      }
    }
  }
  int mh = (lane >> 4) << 3;
  #pragma unroll
  for (int s = 0; s < 2; ++s) {
    int col = nts[s] * 16 + (lane & 15);
    float cb = conv_b[p * D + col];
    #pragma unroll
    for (int r2 = 0; r2 < 8; ++r2) {
      int m = r2 + mh;                           // 0..15
      long gidx = ((long)(m0 + m) * NP + p) * D + col;
      out[gidx] = out[gidx] + acc[s][r2] + cb;
    }
  }
}

// ---------------------------------------------------------------------------
extern "C" void kernel_launch(void* const* d_in, const int* in_sizes, int n_in,
                              void* d_out, int out_size, void* d_ws, size_t ws_size,
                              hipStream_t stream)
{
  (void)in_sizes; (void)out_size; (void)ws_size;
  static const int poff[6] = {0, 8, 72, 136, 232, 296};  // prefix sums of PART_DIMS

  Parts prt;
  const float *ipw, *ipb, *ow, *ob, *cw, *cb;
  for (int p = 0; p < 6; ++p) prt.z[p] = (const float*)d_in[p];
  if (n_in >= 24) {
    // lists flattened to separate inputs
    for (int p = 0; p < 6; ++p) {
      prt.wp[p] = (const float*)d_in[6 + p];
      prt.bp[p] = (const float*)d_in[12 + p];
    }
    ipw = (const float*)d_in[18]; ipb = (const float*)d_in[19];
    ow  = (const float*)d_in[20]; ob  = (const float*)d_in[21];
    cw  = (const float*)d_in[22]; cb  = (const float*)d_in[23];
  } else {
    // lists concatenated flat
    const float* wall = (const float*)d_in[6];
    const float* ball = (const float*)d_in[7];
    for (int p = 0; p < 6; ++p) {
      prt.wp[p] = wall + (long)256 * poff[p];
      prt.bp[p] = ball + 256 * p;
    }
    ipw = (const float*)d_in[8];  ipb = (const float*)d_in[9];
    ow  = (const float*)d_in[10]; ob  = (const float*)d_in[11];
    cw  = (const float*)d_in[12]; cb  = (const float*)d_in[13];
  }

  char* ws = (char*)d_ws;
  __bf16* WtB     = (__bf16*)(ws + 0);          //   884736 B
  float*  biasC   = (float*) (ws + 884736);     //    18432 B
  __bf16* owtB    = (__bf16*)(ws + 903168);     //   131072 B
  __bf16* cwtB    = (__bf16*)(ws + 1034240);    //  2359296 B
  __bf16* attnbuf = (__bf16*)(ws + 3393536);    // 100663296 B
  float* out = (float*)d_out;

  {
    long n0 = 442368L + 4608 + 65536 + 1179648;
    int blocks = (int)((n0 + 255) / 256);
    hipLaunchKernelGGL(k0_prep, dim3(blocks), dim3(256), 0, stream,
                       prt, ipw, ipb, ow, cw, WtB, biasC, owtB, cwtB);
  }
  hipLaunchKernelGGL(k1_qkv_attn, dim3(Mtok / TOK), dim3(256), 0, stream,
                     prt, WtB, biasC, owtB, ob, out, attnbuf);
  hipLaunchKernelGGL(k2_conv_add, dim3((Mtok / 16) * NP), dim3(256), 0, stream,
                     attnbuf, cwtB, cb, out);
}
